// GNNEncoder_32607391711819
// MI455X (gfx1250) — matrix operand order repaired and checked by reading, not compiled, not run on hardware
//
#include <hip/hip_runtime.h>

// GCN 2-layer forward for MI455X (gfx1250, wave32, WMMA).
//
//   deg[d] = 1 + #edges(dst==d); dis = rsqrt(deg)
//   L1: hw = X @ W1 (100000x128 @ 128x64); h1[dst] += dis[s]*dis[d]*hw[src]; +b1; relu
//   L2: hw = h1 @ W2 (100000x64 @ 64x64);  out[dst] += ...;                  +b2; relu
//
// GEMMs: bandwidth-bound -> v_wmma_f32_16x16x32_f16, f32 accumulate.
//   - W pre-packed into fragment-ready layout (one 32B contiguous load per
//     lane per k-step) and held in registers across a grid-stride M loop.
//   - K templated (128/64) so the k-loop fully unrolls into back-to-back WMMAs.
// Edge phase: dominant cost (~1.3GB semi-random traffic). float4/thread so
// each edge is one coalesced 256B gather; scatter via global_atomic_add_f32.

typedef __attribute__((ext_vector_type(16))) _Float16 v16h;
typedef __attribute__((ext_vector_type(8)))  float    v8f;

#define FEAT 64

// ---------- degree / norm ----------
__global__ void k_init_deg(float* deg, int n) {
    int i = blockIdx.x * blockDim.x + threadIdx.x;
    if (i < n) deg[i] = 1.0f;               // self-loop contribution
}

__global__ void k_accum_deg(const long long* __restrict__ dst, float* deg, int nE) {
    int i = blockIdx.x * blockDim.x + threadIdx.x;
    if (i < nE) atomicAdd(&deg[(int)dst[i]], 1.0f);
}

__global__ void k_rsqrt(float* deg, int n) {
    int i = blockIdx.x * blockDim.x + threadIdx.x;
    if (i < n) {
        float d = deg[i];
        deg[i] = (d > 0.0f) ? __frsqrt_rn(d) : 0.0f;   // deg >= 1 always; keep guard
    }
}

// ---------- weight pack: f32 [K][64] -> f16 fragment-ready ----------
// Wp layout: [ks][half][col][j] (j=0..15 contiguous), where the B-fragment
// element j for lane with (half, col) at k-step ks is W[ks*32 + half*16 + j][col].
// Each lane's 16 halfs are then one 32-byte contiguous load.
__global__ void k_pack_w(const float* __restrict__ W, _Float16* __restrict__ Wp, int K) {
    int i = blockIdx.x * blockDim.x + threadIdx.x;
    if (i >= K * FEAT) return;
    int j    = i & 15;
    int col  = (i >> 4) & 63;
    int hs   = i >> 10;            // ks*2 + half
    int half = hs & 1;
    int ks   = hs >> 1;
    int krow = ks * 32 + half * 16 + j;
    Wp[i] = (_Float16)W[(size_t)krow * FEAT + col];
}

// ---------- WMMA GEMM: Y[M x 64] = A[M x K] (f32 -> f16) @ W (packed f16) ----------
// blockDim = 128: wave w owns ntile w (N = 64 = 4 tiles). B fragments loaded
// once into registers, then grid-stride loop over 16-row M tiles.
template <int K>
__global__ void __launch_bounds__(128)
k_gemm_wmma(const float* __restrict__ A, const _Float16* __restrict__ Wp,
            float* __restrict__ Y, int mtiles) {
    constexpr int KSTEPS = K / 32;

    const int lane  = threadIdx.x & 31;
    const int ntile = threadIdx.x >> 5;           // 0..3
    const int half  = lane >> 4;                  // 0: lanes 0-15, 1: lanes 16-31
    const int l15   = lane & 15;
    const int col   = ntile * 16 + l15;           // B/D column
    const int kbA   = half ? 16 : 0;              // A: K in {kbA..kbA+15}

    // Persistent B fragments: one 32B contiguous load per k-step.
    v16h b[KSTEPS];
    #pragma unroll
    for (int ks = 0; ks < KSTEPS; ++ks)
        b[ks] = *(const v16h*)(Wp + (((size_t)(ks * 2 + half) * FEAT + col) << 4));

    for (int mt = blockIdx.x; mt < mtiles; mt += gridDim.x) {
        const int row = mt * 16 + l15;
        const float* xr0 = A + (size_t)row * K + kbA;

        v8f acc = {};
        #pragma unroll
        for (int ks = 0; ks < KSTEPS; ++ks) {
            // A fragment (16x32 f16): lane holds rows row, K runs {0-15}(+kbA).
            v16h a;
            const float* xr = xr0 + ks * 32;
            #pragma unroll
            for (int i = 0; i < 8; ++i) a[i]     = (_Float16)xr[i];
            #pragma unroll
            for (int i = 0; i < 8; ++i) a[8 + i] = (_Float16)xr[8 + i];

            acc = __builtin_amdgcn_wmma_f32_16x16x32_f16(
                false, a, false, b[ks], (short)0, acc, false, false);
        }

        // D layout: VGPR v -> M = v + 8*half; N = col.
        const int mbase = mt * 16 + (half ? 8 : 0);
        #pragma unroll
        for (int v = 0; v < 8; ++v)
            Y[(size_t)(mbase + v) * FEAT + col] = acc[v];
    }
}

// ---------- zero ----------
__global__ void k_zero(float* p, long long n) {
    long long i = (long long)blockIdx.x * blockDim.x + threadIdx.x;
    if (i < n) p[i] = 0.0f;
}

// ---------- edge scatter: out[dst] += dis[src]*dis[dst] * hw[src] ----------
// 16 threads per edge, float4 each: fully coalesced 256B gather per edge.
// Edges [0,nE) real, [nE, nE+nN) self-loops.
__global__ void k_scatter(const float4* __restrict__ hw4,
                          const long long* __restrict__ src,
                          const long long* __restrict__ dst,
                          const float* __restrict__ dis,
                          float* __restrict__ out,
                          int nE, int nN) {
    long long t = (long long)blockIdx.x * blockDim.x + threadIdx.x;
    long long total = (long long)(nE + nN) * 16;
    if (t >= total) return;
    int e = (int)(t >> 4);
    int q = (int)(t & 15);
    int s, d;
    if (e < nE) { s = (int)src[e]; d = (int)dst[e]; }
    else        { s = e - nE;      d = s; }
    float norm = dis[s] * dis[d];
    float4 v = hw4[(size_t)s * 16 + q];
    float* op = out + ((size_t)d * FEAT + q * 4);
    atomicAdd(op + 0, v.x * norm);
    atomicAdd(op + 1, v.y * norm);
    atomicAdd(op + 2, v.z * norm);
    atomicAdd(op + 3, v.w * norm);
}

// ---------- bias + relu (in place) ----------
__global__ void k_bias_relu(float* __restrict__ h, const float* __restrict__ bias, long long n) {
    long long i = (long long)blockIdx.x * blockDim.x + threadIdx.x;
    if (i < n) {
        float v = h[i] + bias[i & (FEAT - 1)];
        h[i] = v > 0.0f ? v : 0.0f;
    }
}

static inline size_t align_up(size_t x, size_t a) { return (x + a - 1) & ~(a - 1); }

extern "C" void kernel_launch(void* const* d_in, const int* in_sizes, int n_in,
                              void* d_out, int out_size, void* d_ws, size_t ws_size,
                              hipStream_t stream) {
    const float*     x    = (const float*)d_in[0];      // [N,128]
    const long long* ei   = (const long long*)d_in[1];  // [2,E] int64
    const float*     W1   = (const float*)d_in[2];      // [128,64]
    const float*     b1   = (const float*)d_in[3];      // [64]
    const float*     W2   = (const float*)d_in[4];      // [64,64]
    const float*     b2   = (const float*)d_in[5];      // [64]
    float*           out  = (float*)d_out;              // [N,64]

    const int IN_DIM = 128;
    const int N = in_sizes[0] / IN_DIM;                 // 100000
    const int E = in_sizes[1] / 2;                      // 1200000
    const long long* srcIdx = ei;
    const long long* dstIdx = ei + E;

    // Workspace layout
    char* ws = (char*)d_ws;
    size_t off = 0;
    float* dis = (float*)(ws + off);       off = align_up(off + (size_t)N * 4, 256);
    _Float16* W1p = (_Float16*)(ws + off); off = align_up(off + (size_t)IN_DIM * FEAT * 2, 256);
    _Float16* W2p = (_Float16*)(ws + off); off = align_up(off + (size_t)FEAT * FEAT * 2, 256);
    float* hw = (float*)(ws + off);        off = align_up(off + (size_t)N * FEAT * 4, 256);
    float* h1 = (float*)(ws + off);        off = align_up(off + (size_t)N * FEAT * 4, 256);
    (void)ws_size;

    const int TB = 256;
    const long long nElem = (long long)N * FEAT;
    const long long nScat = (long long)(E + N) * 16;    // float4 granularity
    const int gN    = (N + TB - 1) / TB;
    const int gE    = (E + TB - 1) / TB;
    const int gElem = (int)((nElem + TB - 1) / TB);
    const int gScat = (int)((nScat + TB - 1) / TB);
    const int mtiles = N / 16;                          // 6250 (exact)
    const int gGemm  = mtiles < 2048 ? mtiles : 2048;   // persistent-B grid-stride

    // degree -> dis = rsqrt(deg)
    k_init_deg<<<gN, TB, 0, stream>>>(dis, N);
    k_accum_deg<<<gE, TB, 0, stream>>>(dstIdx, dis, E);
    k_rsqrt<<<gN, TB, 0, stream>>>(dis, N);

    // weights -> packed f16 fragments
    k_pack_w<<<(IN_DIM * FEAT + TB - 1) / TB, TB, 0, stream>>>(W1, W1p, IN_DIM);
    k_pack_w<<<(FEAT * FEAT + TB - 1) / TB, TB, 0, stream>>>(W2, W2p, FEAT);

    // ---- Layer 1 ----
    k_gemm_wmma<128><<<gGemm, 128, 0, stream>>>(x, W1p, hw, mtiles);
    k_zero<<<gElem, TB, 0, stream>>>(h1, nElem);
    k_scatter<<<gScat, TB, 0, stream>>>((const float4*)hw, srcIdx, dstIdx, dis, h1, E, N);
    k_bias_relu<<<gElem, TB, 0, stream>>>(h1, b1, nElem);

    // ---- Layer 2 ----
    k_gemm_wmma<64><<<gGemm, 128, 0, stream>>>(h1, W2p, hw, mtiles);
    k_zero<<<gElem, TB, 0, stream>>>(out, nElem);
    k_scatter<<<gScat, TB, 0, stream>>>((const float4*)hw, srcIdx, dstIdx, dis, out, E, N);
    k_bias_relu<<<gElem, TB, 0, stream>>>(out, b2, nElem);
}
